// HGraphEncoder_13365938225236
// MI455X (gfx1250) — matrix operand order repaired and checked by reading, not compiled, not run on hardware
//
#include <hip/hip_runtime.h>
#include <math.h>
#include <stdint.h>

// ---------------------------------------------------------------------------
// HANConv 2-layer heterogeneous graph encoder for MI455X (gfx1250, wave32).
// GEMMs: v_wmma_f32_16x16x32_bf16, 4 accumulators/wave (A-fragment reuse).
//   - A tile: global_load_async_to_lds_b128 (ASYNCcnt) -> one bf16 cvt pass.
//   - W: pre-swizzled to bf16 fragment-native layout -> b128 loads, no cvt.
//   - All 4 B fragments staged in distinct registers before the 4 WMMAs so
//     loads overlap the matrix pipe (no per-WMMA s_wait_loadcnt 0x0).
// Edge phase: ordered-uint atomicMax segment-max + fp32 atomic scatter.
// ---------------------------------------------------------------------------

#define N_NODES 50000
#define N_EDGES 256000
#define DIN     128
#define DOUT    256
#define NHEAD   4
#define HDIM    64

typedef __attribute__((ext_vector_type(16))) __bf16 v16bf;
typedef __attribute__((ext_vector_type(8)))  float  v8f;

// -------------------- W swizzle: fp32 row-major -> bf16 fragments ----------
// Wsw[(ks*256 + c)*32 + ki] = bf16( W[(ks*32 + ki)*256 + c] )
// A lane's B fragment (col c, laneHi) = 16 contiguous bf16 at ki = 16*laneHi.
__global__ void swizzle_w_kernel(const float* __restrict__ W,
                                 __bf16* __restrict__ Wsw, int K)
{
    int i = blockIdx.x * blockDim.x + threadIdx.x;   // input-coalesced
    if (i >= K * DOUT) return;
    int k = i >> 8;                                  // K index
    int c = i & 255;                                 // column
    int ks = k >> 5, ki = k & 31;
    Wsw[((size_t)ks * DOUT + c) * 32 + ki] = (__bf16)W[i];
}

// ------------------------------ GEMM (WMMA) --------------------------------
// C[M x 256] = act(A[M x K] @ W[K x 256] + bias), act: 0=none, 1=relu, 2=tanh
// Block = 256 threads = 8 waves computing a 32x256 tile:
//   wave -> (rowHalf = w>>2) * 16 rows, (colGroup = w&3) * 64 cols,
//   4 accumulators per wave; one A fragment feeds 4 WMMAs per K-step.
__global__ __launch_bounds__(256)
void gemm_bias_act(const float* __restrict__ A, const __bf16* __restrict__ Wsw,
                   const float* __restrict__ bias, float* __restrict__ C,
                   int M, int K, int act)
{
    __shared__ float  Asf[32 * 256];             // async fp32 stage (<=32 KB)
    __shared__ __bf16 Asb[32 * 256];             // bf16 copy       (<=16 KB)
    const int tid     = threadIdx.x;
    const int rowBase = blockIdx.x * 32;

    // --- async-stage the 32 x K fp32 A tile into LDS (16 B per lane/iter) ---
    {
        // Low 32 bits of a generic shared pointer = LDS byte offset.
        unsigned ldsBase   = (unsigned)(uintptr_t)&Asf[0];
        const int rowBytes = K * 4;
        const int bytes    = 32 * rowBytes;
        for (int off = tid * 16; off < bytes; off += 256 * 16) {
            int row   = off / rowBytes;
            int inrow = off - row * rowBytes;
            int gr = rowBase + row;
            if (gr >= M) gr = M - 1;             // clamp: stay in-bounds
            const void* gp = (const char*)(A + (size_t)gr * K) + inrow;
            unsigned    lp = ldsBase + off;
            asm volatile("global_load_async_to_lds_b128 %0, %1, off"
                         :: "v"(lp), "v"(gp) : "memory");
        }
        asm volatile("s_wait_asynccnt 0" ::: "memory");
    }
    __syncthreads();

    // --- one-shot fp32 -> bf16 conversion (uniform trip count) ---
    {
        const int total = 32 * K;
        for (int i = tid; i < total; i += 256) Asb[i] = (__bf16)Asf[i];
    }
    __syncthreads();

    const int wave     = tid >> 5;               // 0..7
    const int lane     = tid & 31;
    const int laneLo   = lane & 15;
    const int laneHi   = lane >> 4;              // 0/1
    const int rowHalf  = wave >> 2;              // 0..1  (16-row strips)
    const int colGroup = wave & 3;               // 0..3  (64-col groups)
    const int rloc     = rowHalf * 16 + laneLo;  // local A row (= fragment M)
    const int col0     = colGroup * 64 + laneLo; // tile-t column = col0 + 16t

    v8f acc[4];
    #pragma unroll
    for (int t = 0; t < 4; ++t) {
        float b = bias[col0 + 16 * t];           // bias depends on column only
        #pragma unroll
        for (int r = 0; r < 8; ++r) acc[t][r] = b;
    }

    const __bf16* arow = Asb + (size_t)rloc * K;
    for (int kk = 0; kk < K; kk += 32) {
        const __bf16* wslab = Wsw + (size_t)(kk >> 5) * DOUT * 32;
        if (kk + 32 < K)                         // prefetch next W slab
            __builtin_prefetch(wslab + (size_t)DOUT * 32, 0, 1);

        // A 16x32 bf16 fragment (this code's contiguous k order, same as B): lane = M (mod 16),
        // element j -> K = j + 16*laneHi  => one contiguous 32 B run
        v16bf a;
        #pragma unroll
        for (int j = 0; j < 8; ++j) a[j]     = arow[kk + 16 * laneHi + j];
        #pragma unroll
        for (int j = 0; j < 8; ++j) a[8 + j] = arow[kk + 16 * laneHi + 8 + j];

        // Stage all 4 B fragments in distinct registers (one load clause),
        // then issue 4 back-to-back WMMAs sharing the A fragment.
        v16bf bf0, bf1, bf2, bf3;
        {
            const __bf16* bp0 = wslab + (size_t)(col0 +  0) * 32 + 16 * laneHi;
            const __bf16* bp1 = wslab + (size_t)(col0 + 16) * 32 + 16 * laneHi;
            const __bf16* bp2 = wslab + (size_t)(col0 + 32) * 32 + 16 * laneHi;
            const __bf16* bp3 = wslab + (size_t)(col0 + 48) * 32 + 16 * laneHi;
            #pragma unroll
            for (int j = 0; j < 16; ++j) bf0[j] = bp0[j];
            #pragma unroll
            for (int j = 0; j < 16; ++j) bf1[j] = bp1[j];
            #pragma unroll
            for (int j = 0; j < 16; ++j) bf2[j] = bp2[j];
            #pragma unroll
            for (int j = 0; j < 16; ++j) bf3[j] = bp3[j];
        }
        acc[0] = __builtin_amdgcn_wmma_f32_16x16x32_bf16(
                     false, a, false, bf0, (short)0, acc[0], false, false);
        acc[1] = __builtin_amdgcn_wmma_f32_16x16x32_bf16(
                     false, a, false, bf1, (short)0, acc[1], false, false);
        acc[2] = __builtin_amdgcn_wmma_f32_16x16x32_bf16(
                     false, a, false, bf2, (short)0, acc[2], false, false);
        acc[3] = __builtin_amdgcn_wmma_f32_16x16x32_bf16(
                     false, a, false, bf3, (short)0, acc[3], false, false);
    }

    // D 16x16 f32 layout: VGPR r -> row r + 8*laneHi, col = lane%16
    #pragma unroll
    for (int t = 0; t < 4; ++t) {
        #pragma unroll
        for (int r = 0; r < 8; ++r) {
            int row = rowBase + rowHalf * 16 + r + 8 * laneHi;
            if (row < M) {
                float v = acc[t][r];
                if (act == 1)      v = fmaxf(v, 0.0f);
                else if (act == 2) v = tanhf(v);
                C[(size_t)row * DOUT + col0 + 16 * t] = v;
            }
        }
    }
}

// ------------------------- elementwise / fill ------------------------------
__global__ void fill_f32(float* p, float v, size_t n) {
    size_t i = (size_t)blockIdx.x * blockDim.x + threadIdx.x;
    if (i < n) p[i] = v;
}
__global__ void fill_u32(unsigned* p, unsigned v, size_t n) {
    size_t i = (size_t)blockIdx.x * blockDim.x + threadIdx.x;
    if (i < n) p[i] = v;
}
__global__ void relu_inplace(float* p, size_t n) {
    size_t i = (size_t)blockIdx.x * blockDim.x + threadIdx.x;
    if (i < n) p[i] = fmaxf(p[i], 0.0f);
}

// ------------------------- attention scores -------------------------------
// out[n*4+h] = sum_d xp[n, h*64+d] * att[h*64+d]
__global__ void score_kernel(const float* __restrict__ xp,
                             const float* __restrict__ att,
                             float* __restrict__ out)
{
    int i = blockIdx.x * blockDim.x + threadIdx.x;     // n*H + h
    if (i >= N_NODES * NHEAD) return;
    int h = i & 3, n = i >> 2;
    const float* xr = xp + (size_t)n * DOUT + h * HDIM;
    const float* ar = att + h * HDIM;
    float s = 0.0f;
    #pragma unroll 8
    for (int d = 0; d < HDIM; ++d) s += xr[d] * ar[d];
    out[i] = s;
}

// ------------------------- segment softmax helpers -------------------------
__device__ __forceinline__ float lrelu02(float x) { return x > 0.0f ? x : 0.2f * x; }
__device__ __forceinline__ unsigned enc_f32(float f) {
    unsigned u = __float_as_uint(f);
    return (u & 0x80000000u) ? ~u : (u | 0x80000000u);   // order-preserving
}
__device__ __forceinline__ float dec_f32(unsigned e) {
    unsigned b = (e & 0x80000000u) ? (e & 0x7FFFFFFFu) : ~e;
    return __uint_as_float(b);
}
__device__ __forceinline__ int finite_f32(float m) {
    return ((__float_as_uint(m) >> 23) & 0xFFu) != 0xFFu;
}

__global__ void seg_max_kernel(const float* __restrict__ as_,
                               const float* __restrict__ ad_,
                               const int* __restrict__ es,
                               const int* __restrict__ ed,
                               unsigned* __restrict__ menc)
{
    int i = blockIdx.x * blockDim.x + threadIdx.x;       // e*H + h
    if (i >= N_EDGES * NHEAD) return;
    int h = i & 3, e = i >> 2;
    int s = es[e], d = ed[e];
    float alpha = lrelu02(as_[s * NHEAD + h] + ad_[d * NHEAD + h]);
    atomicMax(&menc[d * NHEAD + h], enc_f32(alpha));
}

__global__ void seg_exp_kernel(const float* __restrict__ as_,
                               const float* __restrict__ ad_,
                               const int* __restrict__ es,
                               const int* __restrict__ ed,
                               const unsigned* __restrict__ menc,
                               float* __restrict__ ex,
                               float* __restrict__ den)
{
    int i = blockIdx.x * blockDim.x + threadIdx.x;
    if (i >= N_EDGES * NHEAD) return;
    int h = i & 3, e = i >> 2;
    int s = es[e], d = ed[e];
    float alpha = lrelu02(as_[s * NHEAD + h] + ad_[d * NHEAD + h]);
    float m = dec_f32(menc[d * NHEAD + h]);
    if (!finite_f32(m)) m = 0.0f;                         // matches reference
    float x = __expf(alpha - m);
    ex[i] = x;
    atomicAdd(&den[d * NHEAD + h], x);
}

// o[ed, :] += (ex/denom)[h] * xp_src[es, :]   (one block per edge, 256 lanes)
__global__ void scatter_kernel(const float* __restrict__ ex,
                               const float* __restrict__ den,
                               const int* __restrict__ es,
                               const int* __restrict__ ed,
                               const float* __restrict__ xs,
                               float* __restrict__ o)
{
    int e = blockIdx.x;
    int d = threadIdx.x;                                  // 0..255
    int s = es[e], t = ed[e];
    int h = d >> 6;
    float coeff = ex[e * NHEAD + h] / (den[t * NHEAD + h] + 1e-16f);
    atomicAdd(&o[(size_t)t * DOUT + d], coeff * xs[(size_t)s * DOUT + d]);
}

// ------------------------- semantic attention ------------------------------
// wout += sum_i T[i] * q[i % 256]   (grid-stride + block reduce + atomic)
__global__ void reduce_q_kernel(const float* __restrict__ T,
                                const float* __restrict__ q,
                                float* wout, size_t total)
{
    __shared__ float sm[256];
    size_t i0 = (size_t)blockIdx.x * blockDim.x + threadIdx.x;
    size_t stride = (size_t)gridDim.x * blockDim.x;
    float acc = 0.0f;
    for (size_t i = i0; i < total; i += stride) acc += T[i] * q[i & 255];
    sm[threadIdx.x] = acc;
    __syncthreads();
    for (int s = 128; s > 0; s >>= 1) {
        if (threadIdx.x < s) sm[threadIdx.x] += sm[threadIdx.x + s];
        __syncthreads();
    }
    if (threadIdx.x == 0) atomicAdd(wout, sm[0]);
}

// out = softmax(w/N)[0]*s0 + softmax(w/N)[1]*s1   (+ optional relu)
__global__ void combine_kernel(const float* __restrict__ s0,
                               const float* __restrict__ s1,
                               const float* __restrict__ wb,
                               float* __restrict__ out,
                               size_t total, int do_relu, float invN)
{
    size_t i = (size_t)blockIdx.x * blockDim.x + threadIdx.x;
    if (i >= total) return;
    float w0 = wb[0] * invN, w1 = wb[1] * invN;
    float mx = fmaxf(w0, w1);
    float e0 = __expf(w0 - mx), e1 = __expf(w1 - mx);
    float inv = 1.0f / (e0 + e1);
    float v = (e0 * inv) * s0[i] + (e1 * inv) * s1[i];
    if (do_relu) v = fmaxf(v, 0.0f);
    out[i] = v;
}

// ---------------------------------------------------------------------------
extern "C" void kernel_launch(void* const* d_in, const int* in_sizes, int n_in,
                              void* d_out, int out_size, void* d_ws, size_t ws_size,
                              hipStream_t stream)
{
    (void)in_sizes; (void)out_size; (void)ws_size;

    const float* x_inst = (const float*)d_in[0];
    const float* x_net  = (const float*)d_in[1];
    const int* ei[4] = { (const int*)d_in[2], (const int*)d_in[3],
                         (const int*)d_in[4], (const int*)d_in[5] };

    // Per-layer params. Leaf order within a layer = jax tree_flatten sorted
    // keys: att_dst_0..3, att_src_0..3, k_W, k_b, proj_inst_W, proj_inst_b,
    // proj_net_W, proj_net_b, q ('W' < 'b' in ASCII).
    struct LayerP {
        const float *att_dst[4], *att_src[4], *kW, *kb, *piW, *pib, *pnW, *pnb, *q;
    } LP[2];

    if (n_in >= 36) {                     // params flattened to separate leaves
        for (int li = 0; li < 2; ++li) {
            void* const* L = d_in + 6 + 15 * li;
            for (int i = 0; i < 4; ++i) LP[li].att_dst[i] = (const float*)L[i];
            for (int i = 0; i < 4; ++i) LP[li].att_src[i] = (const float*)L[4 + i];
            LP[li].kW  = (const float*)L[8];
            LP[li].kb  = (const float*)L[9];
            LP[li].piW = (const float*)L[10];
            LP[li].pib = (const float*)L[11];
            LP[li].pnW = (const float*)L[12];
            LP[li].pnb = (const float*)L[13];
            LP[li].q   = (const float*)L[14];
        }
    } else {                              // params as one concatenated blob
        const float* p = (const float*)d_in[6];
        for (int li = 0; li < 2; ++li) {
            int din = (li == 0) ? DIN : DOUT;
            for (int i = 0; i < 4; ++i) { LP[li].att_dst[i] = p; p += NHEAD * HDIM; }
            for (int i = 0; i < 4; ++i) { LP[li].att_src[i] = p; p += NHEAD * HDIM; }
            LP[li].kW  = p; p += DOUT * DOUT;
            LP[li].kb  = p; p += DOUT;
            LP[li].piW = p; p += din * DOUT;
            LP[li].pib = p; p += DOUT;
            LP[li].pnW = p; p += din * DOUT;
            LP[li].pnb = p; p += DOUT;
            LP[li].q   = p; p += DOUT;
        }
    }

    // ---------------- scratch layout (~470 MB of fp32) ----------------
    float* ws = (float*)d_ws;
    const size_t NB = (size_t)N_NODES * DOUT;            // 12.8M floats
    const size_t NH = (size_t)N_NODES * NHEAD;           // 200k
    const size_t EH = (size_t)N_EDGES * NHEAD;           // 1.024M
    float* XP_I = ws;                                    // projections
    float* XP_N = ws + 1 * NB;
    float* OB[4] = { ws + 2 * NB, ws + 3 * NB, ws + 4 * NB, ws + 5 * NB };
    float* Tb   = ws + 6 * NB;                           // tanh GEMM temp
    float* R1   = ws + 7 * NB;                           // layer-0 out (inst)
    float* R2   = ws + 8 * NB;                           // layer-0 out (net)
    float* AS   = ws + 9 * NB;
    float* AD   = AS + NH;
    float* DEN  = AD + NH;
    float* EX   = DEN + NH;
    float* WB   = EX + EH;                               // semantic logits (2)
    unsigned* MENC = (unsigned*)(WB + 16);               // NH uints
    __bf16* WSW = (__bf16*)(MENC + NH);                  // swizzled W (128 KB)

    // EDGE_META: e0 inst->net, e1 net->inst, e2 inst->inst, e3 net->net
    const int src_is_inst[4] = { 1, 0, 1, 0 };
    const int dst_is_inst[4] = { 0, 1, 1, 0 };

    const dim3 gb((N_NODES + 31) / 32);                  // GEMM 32-row tiles
    const int  BNH = (int)((NH + 255) / 256);
    const int  BEH = (int)((EH + 255) / 256);
    const int  BNB = (int)((NB + 255) / 256);

    for (int li = 0; li < 2; ++li) {
        const int K = (li == 0) ? DIN : DOUT;
        const int BSW = (K * DOUT + 255) / 256;          // swizzle grid
        const float* xi = (li == 0) ? x_inst : R1;
        const float* xn = (li == 0) ? x_net  : R2;

        // per-node-type projections -> XP_I / XP_N  [N, 256]
        swizzle_w_kernel<<<BSW, 256, 0, stream>>>(LP[li].piW, WSW, K);
        gemm_bias_act<<<gb, 256, 0, stream>>>(xi, WSW, LP[li].pib, XP_I, N_NODES, K, 0);
        swizzle_w_kernel<<<BSW, 256, 0, stream>>>(LP[li].pnW, WSW, K);
        gemm_bias_act<<<gb, 256, 0, stream>>>(xn, WSW, LP[li].pnb, XP_N, N_NODES, K, 0);

        // edge phase: gather -> segment softmax -> scatter-accumulate
        for (int e = 0; e < 4; ++e) {
            const float* xs = src_is_inst[e] ? XP_I : XP_N;
            const float* xd = dst_is_inst[e] ? XP_I : XP_N;
            const int* es = ei[e];
            const int* ed = ei[e] + N_EDGES;

            score_kernel<<<BNH, 256, 0, stream>>>(xs, LP[li].att_src[e], AS);
            score_kernel<<<BNH, 256, 0, stream>>>(xd, LP[li].att_dst[e], AD);
            fill_u32<<<BNH, 256, 0, stream>>>(MENC, 0u, NH);
            fill_f32<<<BNH, 256, 0, stream>>>(DEN, 0.0f, NH);
            seg_max_kernel<<<BEH, 256, 0, stream>>>(AS, AD, es, ed, MENC);
            seg_exp_kernel<<<BEH, 256, 0, stream>>>(AS, AD, es, ed, MENC, EX, DEN);
            fill_f32<<<BNB, 256, 0, stream>>>(OB[e], 0.0f, NB);
            scatter_kernel<<<N_EDGES, 256, 0, stream>>>(EX, DEN, es, ed, xs, OB[e]);
            relu_inplace<<<BNB, 256, 0, stream>>>(OB[e], NB);
        }

        // semantic attention: inst uses {O1, O2}, net uses {O0, O3}
        const int BSW2 = (DOUT * DOUT + 255) / 256;
        swizzle_w_kernel<<<BSW2, 256, 0, stream>>>(LP[li].kW, WSW, DOUT);
        for (int nt = 0; nt < 2; ++nt) {
            float* S0 = (nt == 0) ? OB[1] : OB[0];
            float* S1 = (nt == 0) ? OB[2] : OB[3];
            fill_f32<<<1, 32, 0, stream>>>(WB, 0.0f, 2);
            for (int m = 0; m < 2; ++m) {
                float* Sm = (m == 0) ? S0 : S1;
                gemm_bias_act<<<gb, 256, 0, stream>>>(Sm, WSW, LP[li].kb, Tb,
                                                      N_NODES, DOUT, 2 /*tanh*/);
                reduce_q_kernel<<<1024, 256, 0, stream>>>(Tb, LP[li].q, WB + m, NB);
            }
            float* dst = (li == 0) ? ((nt == 0) ? R1 : R2)
                                   : ((float*)d_out + (size_t)nt * NB);
            combine_kernel<<<BNB, 256, 0, stream>>>(S0, S1, WB, dst, NB,
                                                    (li == 0) ? 1 : 0,
                                                    1.0f / (float)N_NODES);
        }
    }
}
